// RelationYOLOX_27848567947355
// MI455X (gfx1250) — matrix operand order repaired and checked by reading, not compiled, not run on hardware
//
#include <hip/hip_runtime.h>

// ---------------------------------------------------------------------------
// RelationYOLOX head for MI455X (gfx1250, wave32).
// WMMA f32_16x16x32_f16 for all matmuls; Tensor Data Mover streams kv tiles
// into double-buffered LDS (tensor_load_to_lds + TENSORcnt pipelining).
// ---------------------------------------------------------------------------

#define HW      10000
#define BATCH   8
#define CH      256
#define NMAIN   600
#define NMAINP  608
#define NREF    300
#define NREFP   320   // kv rows padded to 320 (10 TDM chunks of 32, in-bounds)

typedef _Float16 h16;
typedef __attribute__((ext_vector_type(16))) _Float16 v16h;
typedef __attribute__((ext_vector_type(8)))  float    v8f;
typedef unsigned int v4u __attribute__((ext_vector_type(4)));
typedef unsigned int v8u __attribute__((ext_vector_type(8)));

// A-matrix (16x32 f16) element j of lane-half h -> K offset within 32-chunk.
// ISA 7.12.2: lanes 0-15: VGPR0..3 = K 0..7, VGPR4..7 = K16..23; lanes 16-31: +8.
__device__ __forceinline__ int koffA(int j, int h) {
  int v = j >> 1, p = j & 1;
  int base = (v < 4) ? (2 * v + p) : (16 + 2 * (v - 4) + p);
  return base + 8 * h;
}

// ---------------------------------------------------------------------------
// Top-k via 8-bit radix select on ordered-uint keys, then rank candidates.
// Matches jax.lax.top_k ordering (descending value, ties -> lower index).
// ---------------------------------------------------------------------------
__global__ __launch_bounds__(256) void topk_kernel(const float* __restrict__ obj,
                                                   int k, int* __restrict__ out_idx) {
  __shared__ unsigned keys[HW];
  __shared__ int hist[256];
  __shared__ unsigned s_prefix;
  __shared__ int s_kneed;
  const int b = blockIdx.x;
  const int tid = threadIdx.x;

  for (int i = tid; i < HW; i += blockDim.x) {
    union { float f; unsigned u; } cv;
    cv.f = obj[b * HW + i];
    unsigned u = cv.u;
    keys[i] = (u & 0x80000000u) ? ~u : (u | 0x80000000u);  // order-preserving
  }
  if (tid == 0) { s_prefix = 0u; s_kneed = k; }
  __syncthreads();

  for (int round = 0; round < 4; ++round) {
    const int shift = 24 - 8 * round;
    for (int d = tid; d < 256; d += blockDim.x) hist[d] = 0;
    __syncthreads();
    const unsigned pref = s_prefix;
    for (int i = tid; i < HW; i += blockDim.x) {
      unsigned key = keys[i];
      bool match = (round == 0) || ((key >> (shift + 8)) == (pref >> (shift + 8)));
      if (match) atomicAdd(&hist[(key >> shift) & 255u], 1);
    }
    __syncthreads();
    if (tid == 0) {
      int kn = s_kneed;
      for (int d = 255; d >= 0; --d) {
        int c = hist[d];
        if (kn <= c) { s_prefix = pref | ((unsigned)d << shift); break; }
        kn -= c;
      }
      s_kneed = kn;
    }
    __syncthreads();
  }

  const unsigned kth = s_prefix;  // exact k-th largest key
  for (int i = tid; i < HW; i += blockDim.x) {
    unsigned key = keys[i];
    if (key >= kth) {
      int grt = 0, eqb = 0;
      for (int j = 0; j < HW; ++j) {
        unsigned kj = keys[j];
        grt += (kj > key);
        eqb += ((kj == key) && (j < i));
      }
      int rank = grt + eqb;
      if (rank < k) out_idx[b * k + rank] = i;
    }
  }
}

// ---------------------------------------------------------------------------
// Gather (B,C,H,W) f32 -> padded row-major (B, Npad, 256) f16. Pad rows = 0.
// ---------------------------------------------------------------------------
__global__ void gather_f16(const float* __restrict__ src, const int* __restrict__ idx,
                           int N, int Npad, h16* __restrict__ dst) {
  int i = blockIdx.x * blockDim.x + threadIdx.x;
  int total = BATCH * Npad * CH;
  if (i >= total) return;
  int c = i & (CH - 1);
  int n = (i >> 8) % Npad;
  int b = i / (Npad * CH);
  float v = 0.f;
  if (n < N) {
    int p = idx[b * N + n];
    v = src[((long)b * CH + c) * HW + p];
  }
  dst[i] = (h16)v;
}

// Gather predictions (B,O,H,W) f32 -> (B, 608, O) f32 (pad rows = 0).
__global__ void gather_pred(const float* __restrict__ pred, const int* __restrict__ idx,
                            int O, float* __restrict__ dst) {
  int i = blockIdx.x * blockDim.x + threadIdx.x;
  int total = BATCH * NMAINP * O;
  if (i >= total) return;
  int o = i % O;
  int n = (i / O) % NMAINP;
  int b = i / (NMAINP * O);
  float v = 0.f;
  if (n < NMAIN) {
    int p = idx[b * NMAIN + n];
    v = pred[((long)b * O + o) * HW + p];
  }
  dst[i] = v;
}

// f32 -> f16 elementwise (weights).
__global__ void conv_f16(const float* __restrict__ src, int n, h16* __restrict__ dst) {
  int i = blockIdx.x * blockDim.x + threadIdx.x;
  if (i < n) dst[i] = (h16)src[i];
}

// ---------------------------------------------------------------------------
// Fused relation + head. One wave (32 lanes) per block handles 16 q-rows.
// kv streamed by the Tensor Data Mover into a double-buffered LDS tile:
//   2-D D#: data_size=2B, tensor 256 x 320, tile 256 x 32,
//   LDS pad: +8 DWORDs every 512B  ->  row stride 272 halves.
// Pipelined: issue chunk k+1 before computing chunk k; wait TENSORcnt<=1.
// ---------------------------------------------------------------------------
#define KVSTR   272    // LDS half-stride (produced by TDM pad: 512B + 32B)
#define SSTR    320    // LDS float-stride for the S (P) tile
#define KVROWS  32     // rows per TDM chunk
#define NCHUNK  10     // NREFP / KVROWS
#define KVBUFH  (KVROWS * KVSTR)   // halves per LDS kv buffer

__device__ __forceinline__ void tdm_issue_kv(unsigned lds_off,
                                             const h16* kvb, int chunk) {
  unsigned long long ga =
      (unsigned long long)kvb + (unsigned long long)chunk * KVROWS * CH * 2u;
  v4u g0;
  g0[0] = 1u;                                    // count=1 (valid descriptor)
  g0[1] = lds_off;                               // LDS byte address of tile
  g0[2] = (unsigned)ga;                          // global_addr[31:0]
  g0[3] = (unsigned)(ga >> 32) | (2u << 30);     // global_addr[56:32] | type=2
  v8u g1;
  g1[0] = (1u << 16)                             // data_size = 2 bytes
        | (1u << 20)                             // pad_enable
        | (6u << 22)                             // pad_interval: 512B (code 6)
        | (7u << 25);                            // pad_amount: 8 DWORDs (code 7)
  g1[1] = (unsigned)CH << 16;                    // tensor_dim0 = 256 (lo 16)
  g1[2] = (unsigned)NREFP << 16;                 // tensor_dim1 = 320 (lo 16)
  g1[3] = (unsigned)CH << 16;                    // tile_dim0 = 256
  g1[4] = (unsigned)KVROWS;                      // tile_dim1 = 32, tile_dim2 = 0
  g1[5] = (unsigned)CH;                          // tensor_dim0_stride = 256
  g1[6] = 0u;
  g1[7] = 0u;
  // Drain this wave's LDS reads of the buffer being overwritten (1 iter old).
  asm volatile("s_wait_dscnt 0x0" ::: "memory");
  asm volatile("tensor_load_to_lds %0, %1" :: "s"(g0), "s"(g1) : "memory");
}

__device__ __forceinline__ void tdm_wait_pipe() {   // oldest DMA retired
  asm volatile("s_wait_tensorcnt 0x1" ::: "memory");
}
__device__ __forceinline__ void tdm_wait_all() {
  asm volatile("s_wait_tensorcnt 0x0" ::: "memory");
}

__global__ __launch_bounds__(32) void attn_head_kernel(
    const h16* __restrict__ qf16, const h16* __restrict__ kvf16,
    const h16* __restrict__ wf16, const float* __restrict__ pg,
    const int* __restrict__ idxm,
    const float* __restrict__ roi_cls, const float* __restrict__ roi_loc,
    const float* __restrict__ roi_id,
    const float* __restrict__ b_cls, const float* __restrict__ b_loc,
    const float* __restrict__ b_id,
    float* __restrict__ out) {
  __shared__ h16   kvs[2 * KVBUFH];   // 34816 B: two 32x256 f16 tiles (padded)
  __shared__ float S[16 * SSTR];      // 20480 B: 16 x 320 logits/probs

  const int lane = threadIdx.x;
  const int col  = lane & 15;
  const int h    = lane >> 4;
  const int mtile = blockIdx.x;
  const int mbase = mtile * 16;
  const int z = blockIdx.z;
  const int branch = z >> 3;          // 0..2
  const int b = z & 7;

  const int    O      = (branch == 0) ? 1 : ((branch == 1) ? 4 : 512);
  const int    wrow   = (branch == 0) ? 0 : ((branch == 1) ? 1 : 5);
  const float* bias   = (branch == 0) ? b_cls : ((branch == 1) ? b_loc : b_id);
  const float* feat   = (branch == 0) ? roi_cls : ((branch == 1) ? roi_loc : roi_id);
  const long   outoff = (branch == 0) ? 0 : ((branch == 1) ? 4800 : 24000);
  const long   pgoff  = (branch == 0) ? 0 : ((branch == 1) ? (long)BATCH * NMAINP
                                                           : (long)BATCH * NMAINP * 5);
  const h16* qb  = qf16  + ((long)branch * BATCH + b) * NMAINP * CH;
  const h16* kvb = kvf16 + ((long)branch * BATCH + b) * NREFP * CH;
  const unsigned kvs_lds = (unsigned)(unsigned long long)(&kvs[0]);

  // ---- pipeline prologue: DMA chunk 0 while we fetch q fragments -------
  tdm_issue_kv(kvs_lds, kvb, 0);

  // ---- A fragments of q (16 rows x 256) --------------------------------
  v16h aq[8];
#pragma unroll
  for (int kt = 0; kt < 8; ++kt) {
#pragma unroll
    for (int j = 0; j < 16; ++j)
      aq[kt][j] = qb[(mbase + col) * CH + 32 * kt + 16 * h + j];
  }

  const float scale = 0.0625f;  // 1/sqrt(256)

  // ---- pass 1: logits S = (q @ kv^T) * scale ---------------------------
  for (int chunk = 0; chunk < NCHUNK; ++chunk) {
    const int buf = chunk & 1;
    if (chunk + 1 < NCHUNK) {
      tdm_issue_kv(kvs_lds + (unsigned)((buf ^ 1) * KVBUFH * 2), kvb, chunk + 1);
      tdm_wait_pipe();   // chunk's data has landed
    } else {
      tdm_wait_all();
    }
    const h16* kb = &kvs[buf * KVBUFH];
#pragma unroll
    for (int s = 0; s < 2; ++s) {
      int ntile = chunk * 2 + s;
      if (ntile < 19) {
        v8f acc = {};
#pragma unroll
        for (int kt = 0; kt < 8; ++kt) {
          v16h bf;
#pragma unroll
          for (int j = 0; j < 16; ++j)                  // B: K=16h+j (c), N=col (n)
            bf[j] = kb[(16 * s + col) * KVSTR + 32 * kt + 16 * h + j];
          acc = __builtin_amdgcn_wmma_f32_16x16x32_f16(false, aq[kt], false, bf,
                                                       (short)0, acc, false, false);
        }
#pragma unroll
        for (int r = 0; r < 8; ++r) {                   // D: M=r+8h, N=col
          int m = r + 8 * h;
          int n = ntile * 16 + col;
          S[m * SSTR + n] = (n < NREF) ? acc[r] * scale : -3.0e38f;
        }
      }
    }
  }
  __syncthreads();

  // ---- softmax (lanes 0..15, one row each) -----------------------------
  if (lane < 16) {
    const int m = lane;
    float mx = -3.0e38f;
    for (int n = 0; n < 304; ++n) mx = fmaxf(mx, S[m * SSTR + n]);
    float sum = 0.f;
    for (int n = 0; n < 304; ++n) {
      float e = __expf(S[m * SSTR + n] - mx);
      S[m * SSTR + n] = e;
      sum += e;
    }
    float inv = 1.f / sum;
    for (int n = 0; n < 304; ++n) S[m * SSTR + n] *= inv;
    for (int n = 304; n < SSTR; ++n) S[m * SSTR + n] = 0.f;
  }
  __syncthreads();

  // ---- pass 2: O = P @ kv (P fragment loaded lazily per chunk) ---------
  v8f oacc[16];
#pragma unroll
  for (int ct = 0; ct < 16; ++ct) oacc[ct] = {};

  tdm_issue_kv(kvs_lds, kvb, 0);
  for (int chunk = 0; chunk < NCHUNK; ++chunk) {
    const int buf = chunk & 1;
    if (chunk + 1 < NCHUNK) {
      tdm_issue_kv(kvs_lds + (unsigned)((buf ^ 1) * KVBUFH * 2), kvb, chunk + 1);
      tdm_wait_pipe();
    } else {
      tdm_wait_all();
    }
    const h16* kb = &kvs[buf * KVBUFH];
    v16h apf;                                           // P: 16 rows x 32 n's
#pragma unroll
    for (int j = 0; j < 16; ++j)
      apf[j] = (h16)S[col * SSTR + 32 * chunk + 16 * h + j];
#pragma unroll
    for (int ct = 0; ct < 16; ++ct) {
      v16h bf;
#pragma unroll
      for (int j = 0; j < 16; ++j)                      // B: K=n local, N=c
        bf[j] = kb[(16 * h + j) * KVSTR + 16 * ct + col];
      oacc[ct] = __builtin_amdgcn_wmma_f32_16x16x32_f16(false, apf, false, bf,
                                                        (short)0, oacc[ct], false, false);
    }
  }
  __syncthreads();

  // ---- residual + transpose refined tile through LDS (reuses S) --------
  h16* Rf = (h16*)S;                                    // 16 x 256, stride KVSTR
  const int* idxb = idxm + b * NMAIN;
#pragma unroll
  for (int ct = 0; ct < 16; ++ct) {
#pragma unroll
    for (int r = 0; r < 8; ++r) {
      int m = r + 8 * h;
      int gm = mbase + m;
      int c = ct * 16 + col;
      float resid = 0.f;
      if (gm < NMAIN) {
        int pix = idxb[gm];
        resid = feat[((long)b * CH + c) * HW + pix];
      }
      Rf[m * KVSTR + c] = (h16)(oacc[ct][r] + resid);
    }
  }
  __syncthreads();

  // ---- pass 3: head out = refined @ w^T + bias + gathered pred ---------
  v16h ar[8];
#pragma unroll
  for (int kt = 0; kt < 8; ++kt) {
#pragma unroll
    for (int j = 0; j < 16; ++j)
      ar[kt][j] = Rf[col * KVSTR + 32 * kt + 16 * h + j];
  }

  const int n_ot = (O + 15) >> 4;
  for (int ot = 0; ot < n_ot; ++ot) {
    v8f acc = {};
#pragma unroll
    for (int kt = 0; kt < 8; ++kt) {
      v16h bf;
      const int o = ot * 16 + col;
#pragma unroll
      for (int j = 0; j < 16; ++j)                      // B: K=c, N=o ; w[o][c]
        bf[j] = (o < O) ? wf16[(long)(wrow + o) * CH + 32 * kt + 16 * h + j] : (h16)0.f;
      acc = __builtin_amdgcn_wmma_f32_16x16x32_f16(false, ar[kt], false, bf,
                                                   (short)0, acc, false, false);
    }
#pragma unroll
    for (int r = 0; r < 8; ++r) {
      int m = r + 8 * h;
      int gm = mbase + m;
      int o = ot * 16 + col;
      if (gm < NMAIN && o < O) {
        out[outoff + ((long)b * O + o) * NMAIN + gm] =
            acc[r] + bias[o] + pg[pgoff + ((long)b * NMAINP + gm) * O + o];
      }
    }
  }
}

// ---------------------------------------------------------------------------
extern "C" void kernel_launch(void* const* d_in, const int* in_sizes, int n_in,
                              void* d_out, int out_size, void* d_ws, size_t ws_size,
                              hipStream_t stream) {
  const float* roi_cls  = (const float*)d_in[0];
  const float* roi_loc  = (const float*)d_in[1];
  const float* roi_id   = (const float*)d_in[2];
  const float* pred_cls = (const float*)d_in[3];
  const float* pred_loc = (const float*)d_in[4];
  const float* pred_id  = (const float*)d_in[5];
  const float* obj_main = (const float*)d_in[6];
  const float* roi_clsR = (const float*)d_in[7];
  const float* roi_locR = (const float*)d_in[8];
  const float* roi_idR  = (const float*)d_in[9];
  const float* obj_ref  = (const float*)d_in[10];
  const float* w_cls    = (const float*)d_in[11];
  const float* b_cls    = (const float*)d_in[12];
  const float* w_loc    = (const float*)d_in[13];
  const float* b_loc    = (const float*)d_in[14];
  const float* w_id     = (const float*)d_in[15];
  const float* b_id     = (const float*)d_in[16];
  float* out = (float*)d_out;

  // Workspace partition (all offsets 16B aligned).
  char* w = (char*)d_ws;
  int* idxm = (int*)w;  w += (size_t)BATCH * NMAIN * 4;            // 19200
  int* idxr = (int*)w;  w += (size_t)BATCH * NREF * 4;             //  9600
  h16* qf16 = (h16*)w;  w += (size_t)3 * BATCH * NMAINP * CH * 2;  // 7.47 MB
  h16* kvf16 = (h16*)w; w += (size_t)3 * BATCH * NREFP * CH * 2;   // 3.93 MB
  h16* wf16 = (h16*)w;  w += (size_t)517 * CH * 2;                 // 264 KB
  float* pg = (float*)w;                                           // 10.06 MB

  // 1) top-k selection
  topk_kernel<<<BATCH, 256, 0, stream>>>(obj_main, NMAIN, idxm);
  topk_kernel<<<BATCH, 256, 0, stream>>>(obj_ref, NREF, idxr);

  // 2) gathers + f16 conversion
  const int gq = (BATCH * NMAINP * CH + 255) / 256;
  const int gk = (BATCH * NREFP * CH + 255) / 256;
  gather_f16<<<gq, 256, 0, stream>>>(roi_cls, idxm, NMAIN, NMAINP, qf16 + 0L * BATCH * NMAINP * CH);
  gather_f16<<<gq, 256, 0, stream>>>(roi_loc, idxm, NMAIN, NMAINP, qf16 + 1L * BATCH * NMAINP * CH);
  gather_f16<<<gq, 256, 0, stream>>>(roi_id,  idxm, NMAIN, NMAINP, qf16 + 2L * BATCH * NMAINP * CH);
  gather_f16<<<gk, 256, 0, stream>>>(roi_clsR, idxr, NREF, NREFP, kvf16 + 0L * BATCH * NREFP * CH);
  gather_f16<<<gk, 256, 0, stream>>>(roi_locR, idxr, NREF, NREFP, kvf16 + 1L * BATCH * NREFP * CH);
  gather_f16<<<gk, 256, 0, stream>>>(roi_idR,  idxr, NREF, NREFP, kvf16 + 2L * BATCH * NREFP * CH);

  gather_pred<<<(BATCH * NMAINP * 1 + 255) / 256, 256, 0, stream>>>(pred_cls, idxm, 1, pg);
  gather_pred<<<(BATCH * NMAINP * 4 + 255) / 256, 256, 0, stream>>>(pred_loc, idxm, 4, pg + (long)BATCH * NMAINP);
  gather_pred<<<(BATCH * NMAINP * 512 + 255) / 256, 256, 0, stream>>>(pred_id, idxm, 512, pg + (long)BATCH * NMAINP * 5);

  conv_f16<<<(1 * CH + 255) / 256, 256, 0, stream>>>(w_cls, 1 * CH, wf16);
  conv_f16<<<(4 * CH + 255) / 256, 256, 0, stream>>>(w_loc, 4 * CH, wf16 + 1L * CH);
  conv_f16<<<(512 * CH + 255) / 256, 256, 0, stream>>>(w_id, 512 * CH, wf16 + 5L * CH);

  // 3) fused relation + head (38 m-tiles x 8 batches x 3 branches)
  dim3 grid(NMAINP / 16, 1, 3 * BATCH);
  attn_head_kernel<<<grid, 32, 0, stream>>>(qf16, kvf16, wf16, pg, idxm,
                                            roi_cls, roi_loc, roi_id,
                                            b_cls, b_loc, b_id, out);
}
